// Convolution_48421461295280
// MI455X (gfx1250) — matrix operand order repaired and checked by reading, not compiled, not run on hardware
//
#include <hip/hip_runtime.h>
#include <hip/hip_bf16.h>
#include <stdint.h>

// ---------------------------------------------------------------------------
// Fused equivariant edge convolution for MI455X (gfx1250, wave32, WMMA).
//
//   w[e, :3072] = emb[e, :128] @ (w_emb / sqrt(128))   <-- 94 GFLOP, WMMA bf16
//   six block contractions consumed in-register; segment-mean via f32 atomics.
//
// v2 changes vs v1:
//  * 6 waves / block, each wave = one 16-edge tile; the prepacked weight
//    A-tiles are shared through LDS (6x less L2 traffic).
//  * A-tiles streamed with global_load_async_to_lds_b128 (ASYNCcnt) into a
//    double buffer; s_wait_asynccnt + barrier; prefetch overlaps WMMA.
//  * B operands + epilogue tables built straight from global (x fits in L2),
//    no fp32 emb staging buffer -> static LDS ~54 KB.
//
// WMMA orientation:  D(16x16) = A(16 w-cols x 32 K) x B(32 K x 16 edges)
//   lane l of D holds edge n = l%16 for every VGPR -> per-edge epilogue
//   multipliers are plain per-lane LDS broadcasts; 96 output cols accumulate
//   in C-layout v8f registers.
// Precision: split-bf16 (hi+lo), 3 WMMAs per K-chunk, fp32 accumulate.
//
// Workspace (dwords):
//   [0       , 393216)   A tiles: per mt, 2048 dwords = [hi kk0..3 | lo kk0..3]
//   [393216  , 1545216)  summed: 12000 x 96 f32 atomic accumulators
//   [1545216 , 1557216)  cnt:    12000 f32
// ---------------------------------------------------------------------------

typedef __attribute__((ext_vector_type(16))) __bf16 v16bf;
typedef __attribute__((ext_vector_type(8)))  float  v8f;

#define NNODES 12000
#define NEDGES 120000
#define XROW   80
#define NW     3072
#define MSGC   96
#define WPB    6            // waves per block
#define NMT    192          // 16-column tiles of w

#define WS_SUM   393216
#define WS_CNT   1545216
#define WS_ZERON 1164000

// per-wave epilogue table layout (floats)
#define TB_X1S 0            // x1s[u][e], u<32
#define TB_VD  512          // vdot[u][e] (INV_SQRT3 folded), u<16
#define TB_XV  768          // xv[u*3+k][e] (raw, y0 applied in epilogue)
#define TB_SZ  1536

static constexpr float INV_SQRT3 = 0.5773502691896258f;
static constexpr float SQRT2     = 1.4142135623730951f;
static constexpr float N_0E      = 0.14433756729740643f;   // 1/sqrt(48)
static constexpr float N_1O      = 0.25f;                  // sqrt(3/48)
static constexpr float INV_SQRTK = 0.08838834764831845f;   // 1/sqrt(128)

__device__ __forceinline__ uint32_t pack_bf2(__bf16 a, __bf16 b) {
    union { __bf16 h[2]; uint32_t u; } x;
    x.h[0] = a; x.h[1] = b;
    return x.u;
}

// CDNA5 async global->LDS copy (tracked by ASYNCcnt). GVS addressing:
//   mem = SADDR(64) + VADDR(32) ; LDS dest address from VDST VGPR.
__device__ __forceinline__ void async_cp16(uint32_t lds_addr, uint32_t gl_off,
                                           const uint32_t* base) {
#if defined(__HIP_DEVICE_COMPILE__)
    asm volatile("global_load_async_to_lds_b128 %0, %1, %2"
                 :: "v"(lds_addr), "v"(gl_off), "s"(base)
                 : "memory");
#endif
}
__device__ __forceinline__ void wait_async0() {
#if defined(__HIP_DEVICE_COMPILE__)
#if __has_builtin(__builtin_amdgcn_s_wait_asynccnt)
    __builtin_amdgcn_s_wait_asynccnt(0);
#else
    asm volatile("s_wait_asynccnt 0x0" ::: "memory");
#endif
#endif
}

// --------------------------------------------------------------------------
__global__ void zero_accum(float* __restrict__ p, int n) {
    int i = blockIdx.x * blockDim.x + threadIdx.x;
    if (i < n) p[i] = 0.0f;
}

// --------------------------------------------------------------------------
// Prepack w_emb (128x3072 f32) into split-bf16 WMMA A tiles, 8KB per m-tile:
//   ws[mt*2048 + kk*256 + lane*8 + r]        = hi pair
//   ws[mt*2048 + 1024 + kk*256 + lane*8 + r] = lo pair
// ISA 16-bit A 16x32 layout: lane l -> M=l%16 (w col c = mt*16+l%16),
//   VGPR r, half h -> K = 2*(r&3) + h + 8*(l>>4) + 16*(r>>2).
// Normalization constants folded here.
// --------------------------------------------------------------------------
__global__ void __launch_bounds__(32)
prep_weights(const float* __restrict__ w_emb, uint32_t* __restrict__ ws) {
    const int tile = blockIdx.x;            // 0..767 = mt*4 + kk
    const int lane = threadIdx.x;
    const int mt = tile >> 2, kk = tile & 3;
    const int c  = mt * 16 + (lane & 15);
    const int hl = lane >> 4;
    const float scale = (c < 2304) ? (N_0E * INV_SQRTK)
                                   : (N_1O * INV_SQRT3 * INV_SQRTK);
    uint32_t* hi = ws + (size_t)mt * 2048 + (size_t)kk * 256 + (size_t)lane * 8;
    uint32_t* lo = hi + 1024;
#pragma unroll
    for (int r = 0; r < 8; ++r) {
        int K0 = kk * 32 + 2 * (r & 3) + 8 * hl + 16 * (r >> 2);
        float f0 = w_emb[(size_t)K0 * NW + c] * scale;
        float f1 = w_emb[(size_t)(K0 + 1) * NW + c] * scale;
        __bf16 h0 = (__bf16)f0, h1 = (__bf16)f1;
        __bf16 l0 = (__bf16)(f0 - (float)h0);
        __bf16 l1 = (__bf16)(f1 - (float)h1);
        hi[r] = pack_bf2(h0, h1);
        lo[r] = pack_bf2(l0, l1);
    }
}

// --------------------------------------------------------------------------
// Main fused kernel: 6 waves/block, one 16-edge tile per wave, A shared in LDS.
// --------------------------------------------------------------------------
__global__ void __launch_bounds__(WPB * 32)
edge_kernel(const float* __restrict__ x, const float* __restrict__ edge_attr,
            const float* __restrict__ Yij, const int* __restrict__ edge_index,
            const uint32_t* __restrict__ wsA,
            float* __restrict__ summed, float* __restrict__ cnt) {
    __shared__ uint32_t Abuf[2][2048];      // double-buffered A m-tile (8KB each)
    __shared__ float    tabs[WPB * TB_SZ];  // per-wave epilogue tables
    __shared__ int      eidxS[WPB][32];     // [0..15]=dst, [16..31]=src

    const int tid  = threadIdx.x;
    const int w    = tid >> 5;
    const int lane = tid & 31;
    const int eL   = lane & 15;
    const int hl   = lane >> 4;
    const int eb   = (blockIdx.x * WPB + w) * 16;
    float* tab = tabs + w * TB_SZ;

    if (lane < 16) {
        eidxS[w][lane]      = edge_index[eb + lane];
        eidxS[w][16 + lane] = edge_index[NEDGES + eb + lane];
    }
    // wave-local LDS use: DS ops are in-order within a wave, no barrier needed
    const int dn = eidxS[w][eL];
    const int sn = eidxS[w][16 + eL];

    // Per-lane y0/y1 of this lane's edge (both lane halves hold same edge).
    float4 yv = *(const float4*)(Yij + (size_t)(eb + eL) * 4);
    const float y0l = yv.x, y1a = yv.y, y1b = yv.z, y1c = yv.w;

    // Per-wave epilogue tables (one edge per lane 0..15): x1s, vdot, raw xv.
    if (lane < 16) {
        const int e = lane;
#pragma unroll 4
        for (int u = 0; u < 32; ++u)
            tab[TB_X1S + u * 16 + e] = x[(size_t)sn * XROW + u];
        for (int u = 0; u < 16; ++u) {
            float a = x[(size_t)sn * XROW + 32 + u * 3 + 0];
            float b = x[(size_t)sn * XROW + 32 + u * 3 + 1];
            float c = x[(size_t)sn * XROW + 32 + u * 3 + 2];
            tab[TB_VD + u * 16 + e] = INV_SQRT3 * (a * y1a + b * y1b + c * y1c);
            tab[TB_XV + (u * 3 + 0) * 16 + e] = a;
            tab[TB_XV + (u * 3 + 1) * 16 + e] = b;
            tab[TB_XV + (u * 3 + 2) * 16 + e] = c;
        }
    }

    // B operands (emb of this wave's 16 edges, split-bf16) straight from L2.
    // B layout: lane l -> edge n=l%16; VGPR r holds K = base + 2*(r&3) +
    // 8*(l>>4) + 16*(r>>2) and +1 (same order as the A tiles).
    // q=0 -> xs[dst], q=1 -> xs[src], q=2,3 -> edge_attr (uniform per q).
    v16bf bhi[4], blo[4];
#pragma unroll
    for (int q = 0; q < 4; ++q) {
#pragma unroll
        for (int r = 0; r < 8; ++r) {
            const int k = q * 32 + 2 * (r & 3) + 8 * hl + 16 * (r >> 2);
            const float* src;
            if (k < 32)       src = x + (size_t)dn * XROW + k;
            else if (k < 64)  src = x + (size_t)sn * XROW + (k - 32);
            else              src = edge_attr + (size_t)(eb + eL) * 64 + (k - 64);
            float2 f = *(const float2*)src;
            __bf16 h0 = (__bf16)f.x, h1 = (__bf16)f.y;
            bhi[q][2 * r]     = h0;
            bhi[q][2 * r + 1] = h1;
            blo[q][2 * r]     = (__bf16)(f.x - (float)h0);
            blo[q][2 * r + 1] = (__bf16)(f.y - (float)h1);
        }
    }

    // Async-stage one 8KB A m-tile (512 x 16B chunks) into Abuf[b].
    auto stage = [&](int mt, int b) {
        uint32_t ldsb = (uint32_t)(uintptr_t)(&Abuf[b][0]);
        for (int c = tid; c < 512; c += WPB * 32)
            async_cp16(ldsb + (uint32_t)c * 16u,
                       (uint32_t)mt * 8192u + (uint32_t)c * 16u, wsA);
    };

    // 12 WMMAs for one m-tile: 4 K-chunks x (hi*hi + hi*lo + lo*hi).
    auto computeD = [&](const uint32_t* Ab) -> v8f {
        v8f d = {};
#pragma unroll
        for (int q = 0; q < 4; ++q) {
            const v16bf ah = *(const v16bf*)(Ab + q * 256 + lane * 8);
            const v16bf al = *(const v16bf*)(Ab + 1024 + q * 256 + lane * 8);
            d = __builtin_amdgcn_wmma_f32_16x16x32_bf16(false, ah, false, bhi[q], (short)0, d, false, false);
            d = __builtin_amdgcn_wmma_f32_16x16x32_bf16(false, ah, false, blo[q], (short)0, d, false, false);
            d = __builtin_amdgcn_wmma_f32_16x16x32_bf16(false, al, false, bhi[q], (short)0, d, false, false);
        }
        return d;
    };

    v8f aS0 = {}, aS1 = {}, aG = {}, aT5 = {}, aV0 = {}, aV1 = {}, aV2 = {};

    stage(0, 0);
    for (int mt = 0; mt < NMT; ++mt) {
        wait_async0();        // own async copies of Abuf[mt&1] done
        __syncthreads();      // whole block sees the tile; prev readers done
        if (mt + 1 < NMT) stage(mt + 1, (mt + 1) & 1);

        v8f d = computeD(&Abuf[mt & 1][0]);

        // Uniform epilogue dispatch (mt is wave-uniform; EXEC stays all-1).
        if (mt < 64) {                       // W1(32x32).sy0 -> out_s
            float m = tab[TB_X1S + (mt >> 1) * 16 + eL] * y0l;
            if (mt & 1) aS1 += d * m; else aS0 += d * m;
        } else if (mt < 96) {                // W2(32x16).sy0 -> out_g
            aG += d * (tab[TB_X1S + (mt - 64) * 16 + eL] * y0l);
        } else if (mt < 128) {               // W3(16x32).vdot -> out_s
            float m = tab[TB_VD + ((mt - 96) >> 1) * 16 + eL];
            if (mt & 1) aS1 += d * m; else aS0 += d * m;
        } else if (mt < 144) {               // W4(16x16).vdot -> out_g
            aG += d * tab[TB_VD + (mt - 128) * 16 + eL];
        } else if (mt < 176) {               // W5(32x16).x1s -> t5
            aT5 += d * tab[TB_X1S + (mt - 144) * 16 + eL];
        } else {                             // W6(16x16).(xv*y0) -> out_v
            const int u = mt - 176;
            aV0 += d * (tab[TB_XV + (u * 3 + 0) * 16 + eL] * y0l);
            aV1 += d * (tab[TB_XV + (u * 3 + 1) * 16 + eL] * y0l);
            aV2 += d * (tab[TB_XV + (u * 3 + 2) * 16 + eL] * y0l);
        }
    }
    aV0 += aT5 * y1a;
    aV1 += aT5 * y1b;
    aV2 += aT5 * y1c;

    // Scatter msg = [out_s(32) | out_g(16) | out_v(48)] into segment sums.
    // Lane l: edge = l%16, w-row m = v + 8*(l>>4) -> all 96 cols covered.
    float* srow = summed + (size_t)dn * MSGC;
    const int mb = hl * 8;
#pragma unroll
    for (int v = 0; v < 8; ++v) {
        int m = mb + v;
        atomicAdd(srow + m,              aS0[v]);
        atomicAdd(srow + 16 + m,         aS1[v]);
        atomicAdd(srow + 32 + m,         aG[v]);
        atomicAdd(srow + 48 + 3 * m,     aV0[v]);
        atomicAdd(srow + 48 + 3 * m + 1, aV1[v]);
        atomicAdd(srow + 48 + 3 * m + 2, aV2[v]);
    }
    if (lane < 16) atomicAdd(cnt + dn, 1.0f);
}

// --------------------------------------------------------------------------
// Finalize: mean, ReLU, gating, residual add. out = x + [s(32) | v(48)].
// --------------------------------------------------------------------------
__global__ void finalize_kernel(const float* __restrict__ x,
                                const float* __restrict__ summed,
                                const float* __restrict__ cnt,
                                float* __restrict__ out) {
    int i = blockIdx.x * blockDim.x + threadIdx.x;
    if (i >= NNODES * XROW) return;
    int n = i / XROW, c = i % XROW;
    float inv = 1.0f / fmaxf(cnt[n], 1.0f);
    const float* srow = summed + (size_t)n * MSGC;
    float r;
    if (c < 32) {
        r = SQRT2 * fmaxf(srow[c] * inv, 0.0f);
    } else {
        int cc = c - 32, u = cc / 3, k = cc - 3 * u;
        float g = SQRT2 * fmaxf(srow[32 + u] * inv, 0.0f);
        r = (srow[48 + u * 3 + k] * inv) * g;
    }
    out[i] = x[i] + r;
}

// --------------------------------------------------------------------------
extern "C" void kernel_launch(void* const* d_in, const int* in_sizes, int n_in,
                              void* d_out, int out_size, void* d_ws, size_t ws_size,
                              hipStream_t stream) {
    const float* x     = (const float*)d_in[0];
    const float* eattr = (const float*)d_in[1];
    const float* Yij   = (const float*)d_in[2];
    const int*   eidx  = (const int*)d_in[3];   // (2,E) int32: dst row, src row
    const float* w_emb = (const float*)d_in[4];

    uint32_t* ws     = (uint32_t*)d_ws;
    float*    summed = (float*)(ws + WS_SUM);
    float*    cntp   = (float*)(ws + WS_CNT);

    zero_accum<<<(WS_ZERON + 255) / 256, 256, 0, stream>>>(summed, WS_ZERON);
    prep_weights<<<768, 32, 0, stream>>>(w_emb, ws);
    edge_kernel<<<NEDGES / (WPB * 16), WPB * 32, 0, stream>>>(
        x, eattr, Yij, eidx, ws, summed, cntp);
    finalize_kernel<<<(NNODES * XROW + 255) / 256, 256, 0, stream>>>(
        x, summed, cntp, (float*)d_out);
}
